// SimpleViTFT_86139864088968
// MI455X (gfx1250) — matrix operand order repaired and checked by reading, not compiled, not run on hardware
//
#include <hip/hip_runtime.h>

typedef _Float16 f16;
typedef __attribute__((ext_vector_type(8)))  f16   v8h;
typedef __attribute__((ext_vector_type(16))) f16   v16h;
typedef __attribute__((ext_vector_type(8)))  float v8f;

#define BATCH  32
#define CCH    3
#define HW     384
#define PP     16
#define DIM    384
#define DEPTH  12
#define NCLS   1000
#define PD     768      // patch dim = 3*16*16
#define NPATCH 576      // 24*24
#define NTOK   577
#define NTOKP  608      // padded to multiple of 32
#define EPSV   1e-5f

// ---------------------------------------------------------------------------
// Weight conversion: wq/wk -> f16 transposed [l][n][k]; Wp -> WpT[n][k]
// ---------------------------------------------------------------------------
__global__ void convert_w_kernel(const float* __restrict__ wq,
                                 const float* __restrict__ wk,
                                 const float* __restrict__ Wp,
                                 f16* __restrict__ wqT, f16* __restrict__ wkT,
                                 f16* __restrict__ WpT) {
  const int DD = DIM * DIM;
  const int total = 2 * DEPTH * DD + PD * DIM;
  for (int idx = blockIdx.x * blockDim.x + threadIdx.x; idx < total;
       idx += gridDim.x * blockDim.x) {
    if (idx < DEPTH * DD) {
      int l = idx / DD, r = idx % DD, n = r / DIM, k = r % DIM;
      wqT[idx] = (f16)wq[((size_t)l * DIM + k) * DIM + n];
    } else if (idx < 2 * DEPTH * DD) {
      int j = idx - DEPTH * DD;
      int l = j / DD, r = j % DD, n = r / DIM, k = r % DIM;
      wkT[j] = (f16)wk[((size_t)l * DIM + k) * DIM + n];
    } else {
      int j = idx - 2 * DEPTH * DD;
      int n = j / PD, k = j % PD;
      WpT[j] = (f16)Wp[(size_t)k * DIM + n];
    }
  }
}

// ---------------------------------------------------------------------------
// wv = (softmax(lv, axis=0)/DIM).T  ->  Bt layout [l][d][e] = sm[d][e]
// ---------------------------------------------------------------------------
__global__ void prep_wv_kernel(const float* __restrict__ lv, f16* __restrict__ wvhT) {
  __shared__ float red[128];
  int e = blockIdx.x, l = blockIdx.y, tid = threadIdx.x;
  float v[3];
  float mx = -3.4e38f;
#pragma unroll
  for (int r = 0; r < 3; r++) {
    int d = tid + 128 * r;
    v[r] = lv[((size_t)l * DIM + d) * DIM + e];
    mx = fmaxf(mx, v[r]);
  }
  red[tid] = mx; __syncthreads();
  for (int s = 64; s > 0; s >>= 1) { if (tid < s) red[tid] = fmaxf(red[tid], red[tid + s]); __syncthreads(); }
  mx = red[0]; __syncthreads();
  float ex[3], sum = 0.f;
#pragma unroll
  for (int r = 0; r < 3; r++) { ex[r] = expf(v[r] - mx); sum += ex[r]; }
  red[tid] = sum; __syncthreads();
  for (int s = 64; s > 0; s >>= 1) { if (tid < s) red[tid] += red[tid + s]; __syncthreads(); }
  float inv = 1.0f / (red[0] * (float)DIM);
#pragma unroll
  for (int r = 0; r < 3; r++) {
    int d = tid + 128 * r;
    wvhT[((size_t)l * DIM + d) * DIM + e] = (f16)(ex[r] * inv);
  }
}

// ---------------------------------------------------------------------------
// Patchify + first LayerNorm -> f16 staging [b][patch][768]
// ---------------------------------------------------------------------------
__global__ void patchify_ln_kernel(const float* __restrict__ img,
                                   const float* __restrict__ g1,
                                   const float* __restrict__ b1,
                                   f16* __restrict__ ph) {
  __shared__ float red[256];
  int bpi = blockIdx.x;
  int b = bpi / NPATCH, p = bpi % NPATCH;
  int py = p / 24, px = p % 24;
  int tid = threadIdx.x;
  float v[3];
  float sum = 0.f;
#pragma unroll
  for (int r = 0; r < 3; r++) {
    int d = tid + 256 * r;
    int pr = d / 48, rem = d % 48, pc = rem / 3, c = rem % 3;
    v[r] = img[(((size_t)b * CCH + c) * HW + (py * PP + pr)) * HW + (px * PP + pc)];
    sum += v[r];
  }
  red[tid] = sum; __syncthreads();
  for (int s = 128; s > 0; s >>= 1) { if (tid < s) red[tid] += red[tid + s]; __syncthreads(); }
  float mu = red[0] * (1.0f / PD); __syncthreads();
  float sq = 0.f;
#pragma unroll
  for (int r = 0; r < 3; r++) { float d0 = v[r] - mu; sq += d0 * d0; }
  red[tid] = sq; __syncthreads();
  for (int s = 128; s > 0; s >>= 1) { if (tid < s) red[tid] += red[tid + s]; __syncthreads(); }
  float rs = rsqrtf(red[0] * (1.0f / PD) + EPSV);
  f16* orow = ph + (size_t)bpi * PD;
#pragma unroll
  for (int r = 0; r < 3; r++) {
    int d = tid + 256 * r;
    orow[d] = (f16)((v[r] - mu) * rs * g1[d] + b1[d]);
  }
}

// ---------------------------------------------------------------------------
// WMMA fragment helpers (CDNA5 ISA 7.12.2, wave32):
//   A lane: m = lane&15, half = lane>>4, element e -> k = 16*half + e (this code's order; ISA table differs)
//   B lane: n = lane&15, half = lane>>4, element e -> k = 16*half + e (this code's order; ISA table differs)
//   D lane: n = lane&15, reg r -> m = r + 8*half
// ---------------------------------------------------------------------------
__device__ __forceinline__ v16h load_a_frag(const f16* p) {
  v8h lo = *(const v8h*)(p);         // k = +0 .. +7   (within this half's set)
  v8h hi = *(const v8h*)(p + 8);     // k = +8 .. +15
  v16h a;
#pragma unroll
  for (int i = 0; i < 8; i++) { a[i] = lo[i]; a[8 + i] = hi[i]; }
  return a;
}

__device__ __forceinline__ v8f wmma32(v16h a, v16h b, v8f c) {
  return __builtin_amdgcn_wmma_f32_16x16x32_f16(false, a, false, b,
                                                (short)0, c, false, false);
}

struct Acc4 { v8f c00, c01, c10, c11; };

// One wave computes a 32x32 output tile: 2x2 register tiling, 4 WMMA per
// K-step off 2 A-fragments + 2 B-fragments (halves the loads per WMMA).
__device__ __forceinline__ Acc4 wmma_gemm_tile_2x2(const f16* __restrict__ A, int lda,
                                                   const f16* __restrict__ Bt, int ldb,
                                                   int m0, int n0, int K) {
  int lane = threadIdx.x & 31;
  int hsel = lane >> 4;
  int lr   = lane & 15;
  const f16* arow0 = A  + (size_t)(m0 + lr) * lda + 16 * hsel;
  const f16* arow1 = arow0 + (size_t)16 * lda;
  const f16* brow0 = Bt + (size_t)(n0 + lr) * ldb + 16 * hsel;
  const f16* brow1 = brow0 + (size_t)16 * ldb;
  v8f z = {0.f, 0.f, 0.f, 0.f, 0.f, 0.f, 0.f, 0.f};
  Acc4 c; c.c00 = z; c.c01 = z; c.c10 = z; c.c11 = z;
  for (int k0 = 0; k0 < K; k0 += 32) {
    v16h a0 = load_a_frag(arow0 + k0);
    v16h a1 = load_a_frag(arow1 + k0);
    v16h b0 = *(const v16h*)(brow0 + k0);
    v16h b1 = *(const v16h*)(brow1 + k0);
    c.c00 = wmma32(a0, b0, c.c00);
    c.c01 = wmma32(a0, b1, c.c01);
    c.c10 = wmma32(a1, b0, c.c10);
    c.c11 = wmma32(a1, b1, c.c11);
  }
  return c;
}

// ---------------------------------------------------------------------------
// Patch embedding GEMM: emb = phLN @ Wp + bp   (M=576, N=384, K=768)
// ---------------------------------------------------------------------------
__global__ void patch_gemm_kernel(const f16* __restrict__ ph, const f16* __restrict__ WpT,
                                  const float* __restrict__ bp, float* __restrict__ emb) {
  int wave = threadIdx.x >> 5;
  int n0 = (blockIdx.x * 4 + wave) * 32;
  int m0 = blockIdx.y * 32;
  int b  = blockIdx.z;
  const f16* A = ph + (size_t)b * NPATCH * PD;
  Acc4 c = wmma_gemm_tile_2x2(A, PD, WpT, PD, m0, n0, PD);
  int lane = threadIdx.x & 31, hsel = lane >> 4, nc = lane & 15;
  float bias0 = bp[n0 + nc], bias1 = bp[n0 + 16 + nc];
  float* O = emb + (size_t)b * NPATCH * DIM;
#pragma unroll
  for (int r = 0; r < 8; r++) {
    int m = m0 + r + 8 * hsel;
    O[(size_t)m * DIM + n0 + nc]            = c.c00[r] + bias0;
    O[(size_t)m * DIM + n0 + 16 + nc]       = c.c01[r] + bias1;
    O[(size_t)(m + 16) * DIM + n0 + nc]     = c.c10[r] + bias0;
    O[(size_t)(m + 16) * DIM + n0 + 16 + nc]= c.c11[r] + bias1;
  }
}

// ---------------------------------------------------------------------------
// Second LayerNorm + cls token + pos embed -> x (f32 residual + f16 copy)
// ---------------------------------------------------------------------------
__global__ void finalize_tokens_kernel(const float* __restrict__ emb,
                                       const float* __restrict__ cls,
                                       const float* __restrict__ pos,
                                       const float* __restrict__ g2,
                                       const float* __restrict__ b2,
                                       float* __restrict__ xf32, f16* __restrict__ xh) {
  __shared__ float red[128];
  int t = blockIdx.x, b = blockIdx.y, tid = threadIdx.x;
  float* xo = xf32 + ((size_t)b * NTOKP + t) * DIM;
  f16*   ho = xh   + ((size_t)b * NTOKP + t) * DIM;
  if (t == 0) {
#pragma unroll
    for (int r = 0; r < 3; r++) {
      int d = tid + 128 * r;
      float v = cls[d] + pos[d];
      xo[d] = v; ho[d] = (f16)v;
    }
    return;
  }
  if (t >= NTOK) {  // zero pad rows so GEMMs over padded tokens stay zero
#pragma unroll
    for (int r = 0; r < 3; r++) { int d = tid + 128 * r; xo[d] = 0.f; ho[d] = (f16)0.f; }
    return;
  }
  const float* e = emb + ((size_t)b * NPATCH + (t - 1)) * DIM;
  float v[3], sum = 0.f;
#pragma unroll
  for (int r = 0; r < 3; r++) { int d = tid + 128 * r; v[r] = e[d]; sum += v[r]; }
  red[tid] = sum; __syncthreads();
  for (int s = 64; s > 0; s >>= 1) { if (tid < s) red[tid] += red[tid + s]; __syncthreads(); }
  float mu = red[0] * (1.0f / DIM); __syncthreads();
  float sq = 0.f;
#pragma unroll
  for (int r = 0; r < 3; r++) { float d0 = v[r] - mu; sq += d0 * d0; }
  red[tid] = sq; __syncthreads();
  for (int s = 64; s > 0; s >>= 1) { if (tid < s) red[tid] += red[tid + s]; __syncthreads(); }
  float rs = rsqrtf(red[0] * (1.0f / DIM) + EPSV);
#pragma unroll
  for (int r = 0; r < 3; r++) {
    int d = tid + 128 * r;
    float val = (v[r] - mu) * rs * g2[d] + b2[d] + pos[(size_t)t * DIM + d];
    xo[d] = val; ho[d] = (f16)val;
  }
}

// ---------------------------------------------------------------------------
// q/k/v projections. which = blockIdx.z%3: 0->q, 1->k, 2->v (v written [d][t])
// ---------------------------------------------------------------------------
__global__ void qkv_kernel(const f16* __restrict__ xh,
                           const f16* __restrict__ BtQ, const f16* __restrict__ BtK,
                           const f16* __restrict__ BtV,
                           f16* __restrict__ qh, f16* __restrict__ kh,
                           f16* __restrict__ vT) {
  int wave = threadIdx.x >> 5;
  int n0 = (blockIdx.x * 4 + wave) * 32;   // DIM=384 -> grid.x=3
  int m0 = blockIdx.y * 32;                // NTOKP=608 -> grid.y=19
  int b = blockIdx.z / 3, which = blockIdx.z % 3;
  const f16* A  = xh + (size_t)b * NTOKP * DIM;
  const f16* Bt = (which == 0) ? BtQ : (which == 1) ? BtK : BtV;
  Acc4 c = wmma_gemm_tile_2x2(A, DIM, Bt, DIM, m0, n0, DIM);
  int lane = threadIdx.x & 31, hsel = lane >> 4, nc = lane & 15;
  if (which < 2) {
    f16* O = ((which == 0) ? qh : kh) + (size_t)b * NTOKP * DIM;
#pragma unroll
    for (int r = 0; r < 8; r++) {
      int m = m0 + r + 8 * hsel;
      O[(size_t)m * DIM + n0 + nc]             = (f16)c.c00[r];
      O[(size_t)m * DIM + n0 + 16 + nc]        = (f16)c.c01[r];
      O[(size_t)(m + 16) * DIM + n0 + nc]      = (f16)c.c10[r];
      O[(size_t)(m + 16) * DIM + n0 + 16 + nc] = (f16)c.c11[r];
    }
  } else {
    f16* O = vT + (size_t)b * DIM * NTOKP;  // transposed: [dim][token]
#pragma unroll
    for (int r = 0; r < 8; r++) {
      int m = m0 + r + 8 * hsel;
      O[(size_t)(n0 + nc) * NTOKP + m]            = (f16)c.c00[r];
      O[(size_t)(n0 + 16 + nc) * NTOKP + m]       = (f16)c.c01[r];
      O[(size_t)(n0 + nc) * NTOKP + m + 16]       = (f16)c.c10[r];
      O[(size_t)(n0 + 16 + nc) * NTOKP + m + 16]  = (f16)c.c11[r];
    }
  }
}

// ---------------------------------------------------------------------------
// scores[b][i][j] = q_i . k_j   (M=N=608, K=384)
// ---------------------------------------------------------------------------
__global__ void scores_kernel(const f16* __restrict__ qh, const f16* __restrict__ kh,
                              float* __restrict__ scoresF) {
  int wave = threadIdx.x >> 5;
  int nt = blockIdx.x * 4 + wave;          // 32-wide N tiles: 19 of 20
  if (nt >= NTOKP / 32) return;            // uniform per-wave exit, no block sync
  int n0 = nt * 32, m0 = blockIdx.y * 32, b = blockIdx.z;
  const f16* A  = qh + (size_t)b * NTOKP * DIM;
  const f16* Bt = kh + (size_t)b * NTOKP * DIM;
  Acc4 c = wmma_gemm_tile_2x2(A, DIM, Bt, DIM, m0, n0, DIM);
  int lane = threadIdx.x & 31, hsel = lane >> 4, nc = lane & 15;
  float* O = scoresF + (size_t)b * NTOKP * NTOKP;
#pragma unroll
  for (int r = 0; r < 8; r++) {
    int m = m0 + r + 8 * hsel;
    O[(size_t)m * NTOKP + n0 + nc]             = c.c00[r];
    O[(size_t)m * NTOKP + n0 + 16 + nc]        = c.c01[r];
    O[(size_t)(m + 16) * NTOKP + n0 + nc]      = c.c10[r];
    O[(size_t)(m + 16) * NTOKP + n0 + 16 + nc] = c.c11[r];
  }
}

// ---------------------------------------------------------------------------
// softmax over axis=1 (over i, per column j), minus identity, -> f16
// one thread per (b, j); consecutive j -> coalesced column walks
// ---------------------------------------------------------------------------
__global__ void attn_softmax_kernel(const float* __restrict__ scoresF,
                                    f16* __restrict__ attnH) {
  int idx = blockIdx.x * blockDim.x + threadIdx.x;
  if (idx >= BATCH * NTOKP) return;
  int b = idx / NTOKP, j = idx % NTOKP;
  const float scale = rsqrtf((float)NTOK);
  f16* A = attnH + (size_t)b * NTOKP * NTOKP + j;
  if (j >= NTOK) {
    for (int i = 0; i < NTOKP; i++) A[(size_t)i * NTOKP] = (f16)0.f;
    return;
  }
  const float* S = scoresF + (size_t)b * NTOKP * NTOKP + j;
  float mx = -3.4e38f;
  for (int i = 0; i < NTOK; i++) mx = fmaxf(mx, S[(size_t)i * NTOKP] * scale);
  float sum = 0.f;
  for (int i = 0; i < NTOK; i++) sum += expf(S[(size_t)i * NTOKP] * scale - mx);
  float inv = 1.0f / sum;
  for (int i = 0; i < NTOKP; i++) {
    float a = 0.f;
    if (i < NTOK)
      a = expf(S[(size_t)i * NTOKP] * scale - mx) * inv - ((i == j) ? 1.0f : 0.0f);
    A[(size_t)i * NTOKP] = (f16)a;
  }
}

// ---------------------------------------------------------------------------
// out = attn @ v ; x = 0.5*out + 0.5*x  (K = 608)
// ---------------------------------------------------------------------------
__global__ void attnv_kernel(const f16* __restrict__ attnH, const f16* __restrict__ vT,
                             float* __restrict__ xf32, f16* __restrict__ xh) {
  int wave = threadIdx.x >> 5;
  int n0 = (blockIdx.x * 4 + wave) * 32;
  int m0 = blockIdx.y * 32;
  int b = blockIdx.z;
  const f16* A  = attnH + (size_t)b * NTOKP * NTOKP;
  const f16* Bt = vT    + (size_t)b * DIM * NTOKP;
  Acc4 c = wmma_gemm_tile_2x2(A, NTOKP, Bt, NTOKP, m0, n0, NTOKP);
  int lane = threadIdx.x & 31, hsel = lane >> 4, nc = lane & 15;
#pragma unroll
  for (int r = 0; r < 8; r++) {
    int m = m0 + r + 8 * hsel;
    size_t o00 = ((size_t)b * NTOKP + m) * DIM + n0 + nc;
    size_t o01 = ((size_t)b * NTOKP + m) * DIM + n0 + 16 + nc;
    size_t o10 = ((size_t)b * NTOKP + m + 16) * DIM + n0 + nc;
    size_t o11 = ((size_t)b * NTOKP + m + 16) * DIM + n0 + 16 + nc;
    float x00 = 0.5f * c.c00[r] + 0.5f * xf32[o00];
    float x01 = 0.5f * c.c01[r] + 0.5f * xf32[o01];
    float x10 = 0.5f * c.c10[r] + 0.5f * xf32[o10];
    float x11 = 0.5f * c.c11[r] + 0.5f * xf32[o11];
    xf32[o00] = x00; xh[o00] = (f16)x00;
    xf32[o01] = x01; xh[o01] = (f16)x01;
    xf32[o10] = x10; xh[o10] = (f16)x10;
    xf32[o11] = x11; xh[o11] = (f16)x11;
  }
}

// ---------------------------------------------------------------------------
// Head: out[b][n] = x[b,0,:] . Wh[:,n] + bh[n]  (f32, tiny)
// ---------------------------------------------------------------------------
__global__ void head_kernel(const float* __restrict__ xf32, const float* __restrict__ Wh,
                            const float* __restrict__ bh, float* __restrict__ out) {
  int n = blockIdx.x * blockDim.x + threadIdx.x;
  int b = blockIdx.y;
  if (n >= NCLS) return;
  const float* xr = xf32 + (size_t)b * NTOKP * DIM;  // token 0 row
  float acc = bh[n];
  for (int d = 0; d < DIM; d++) acc += xr[d] * Wh[(size_t)d * NCLS + n];
  out[(size_t)b * NCLS + n] = acc;
}

// ---------------------------------------------------------------------------
extern "C" void kernel_launch(void* const* d_in, const int* in_sizes, int n_in,
                              void* d_out, int out_size, void* d_ws, size_t ws_size,
                              hipStream_t stream) {
  const float* img   = (const float*)d_in[0];
  const float* ln1_g = (const float*)d_in[1];
  const float* ln1_b = (const float*)d_in[2];
  const float* Wp    = (const float*)d_in[3];
  const float* bp    = (const float*)d_in[4];
  const float* ln2_g = (const float*)d_in[5];
  const float* ln2_b = (const float*)d_in[6];
  const float* pos   = (const float*)d_in[7];
  const float* cls   = (const float*)d_in[8];
  const float* wk    = (const float*)d_in[9];
  const float* wq    = (const float*)d_in[10];
  const float* wv    = (const float*)d_in[11];
  const float* Wh    = (const float*)d_in[12];
  const float* bh    = (const float*)d_in[13];
  float* out = (float*)d_out;

  char* ws = (char*)d_ws;
  size_t off = 0;
  auto take = [&](size_t bytes) -> char* {
    char* p = ws + off;
    off = (off + bytes + 255) & ~(size_t)255;
    return p;
  };
  float* xf32    = (float*)take((size_t)BATCH * NTOKP * DIM * 4);
  f16*   xh      = (f16*)  take((size_t)BATCH * NTOKP * DIM * 2);
  f16*   qh      = (f16*)  take((size_t)BATCH * NTOKP * DIM * 2);
  f16*   kh      = (f16*)  take((size_t)BATCH * NTOKP * DIM * 2);
  f16*   vT      = (f16*)  take((size_t)BATCH * DIM * NTOKP * 2);
  float* scoresF = (float*)take((size_t)BATCH * NTOKP * NTOKP * 4);   // also emb
  char*  shreg   =         take((size_t)BATCH * NPATCH * PD * 2);     // ph / attnH share
  f16*   wqT     = (f16*)  take((size_t)DEPTH * DIM * DIM * 2);
  f16*   wkT     = (f16*)  take((size_t)DEPTH * DIM * DIM * 2);
  f16*   wvhT    = (f16*)  take((size_t)DEPTH * DIM * DIM * 2);
  f16*   WpT     = (f16*)  take((size_t)PD * DIM * 2);
  f16*   attnH   = (f16*)shreg;
  f16*   ph      = (f16*)shreg;
  float* emb     = scoresF;  // patch-embed output lives in scores buffer pre-layers

  convert_w_kernel<<<512, 256, 0, stream>>>(wq, wk, Wp, wqT, wkT, WpT);
  prep_wv_kernel<<<dim3(DIM, DEPTH), 128, 0, stream>>>(wv, wvhT);
  patchify_ln_kernel<<<BATCH * NPATCH, 256, 0, stream>>>(img, ln1_g, ln1_b, ph);
  patch_gemm_kernel<<<dim3(DIM / 128, NPATCH / 32, BATCH), 128, 0, stream>>>(ph, WpT, bp, emb);
  finalize_tokens_kernel<<<dim3(NTOKP, BATCH), 128, 0, stream>>>(emb, cls, pos, ln2_g,
                                                                 ln2_b, xf32, xh);
  for (int l = 0; l < DEPTH; l++) {
    const f16* bq = wqT  + (size_t)l * DIM * DIM;
    const f16* bk = wkT  + (size_t)l * DIM * DIM;
    const f16* bv = wvhT + (size_t)l * DIM * DIM;
    qkv_kernel<<<dim3(DIM / 128, NTOKP / 32, BATCH * 3), 128, 0, stream>>>(
        xh, bq, bk, bv, qh, kh, vT);
    scores_kernel<<<dim3((NTOKP / 32 + 3) / 4, NTOKP / 32, BATCH), 128, 0, stream>>>(
        qh, kh, scoresF);
    attn_softmax_kernel<<<(BATCH * NTOKP + 255) / 256, 256, 0, stream>>>(scoresF, attnH);
    attnv_kernel<<<dim3(DIM / 128, NTOKP / 32, BATCH), 128, 0, stream>>>(attnH, vT, xf32, xh);
  }
  head_kernel<<<dim3((NCLS + 127) / 128, BATCH), 128, 0, stream>>>(xf32, Wh, bh, out);
}
